// SparseLinear_25855703122393
// MI455X (gfx1250) — matrix operand order repaired and checked by reading, not compiled, not run on hardware
//
#include <hip/hip_runtime.h>

// ---------------------------------------------------------------------------
// SparseLinear: y = x @ W^T, W built by scatter-adding COO weights.
// Strategy: densify W (f32 atomics) -> convert W,x to bf16 -> tiled WMMA GEMM
// with double-buffered LDS and gfx1250 async global->LDS loads.
// MI455X / gfx1250: wave32, v_wmma_f32_16x16x32_bf16, 40KB LDS per block.
// ---------------------------------------------------------------------------

typedef __bf16 bf16;
typedef __attribute__((ext_vector_type(4)))  bf16  v4bf;
typedef __attribute__((ext_vector_type(8)))  bf16  v8bf;
typedef __attribute__((ext_vector_type(16))) bf16  v16bf;
typedef __attribute__((ext_vector_type(8)))  float v8f;
typedef __attribute__((ext_vector_type(4)))  int   v4i;

#define IN_F   4096
#define OUT_F  4096
#define BATCH  4096

#define BM 128        // batch rows per block
#define BN 128        // out cols per block
#define BK 32         // K per step (one bf16 WMMA)
#define LDSS 40       // padded LDS row stride in bf16 elements (32 + 8 pad)

#if defined(__gfx1250__) && __has_builtin(__builtin_amdgcn_global_load_async_to_lds_b128)
#define HAVE_ASYNC_LDS 1
#else
#define HAVE_ASYNC_LDS 0
#endif

#if HAVE_ASYNC_LDS
typedef __attribute__((address_space(1))) v4i gas_v4i;   // global
typedef __attribute__((address_space(3))) v4i las_v4i;   // LDS

__device__ __forceinline__ void async_cp16(const bf16* g, bf16* l) {
  __builtin_amdgcn_global_load_async_to_lds_b128(
      (gas_v4i*)(g), (las_v4i*)(l), /*offset=*/0, /*cpol=*/0);
}
__device__ __forceinline__ void wait_async0() {
#if __has_builtin(__builtin_amdgcn_s_wait_asynccnt)
  __builtin_amdgcn_s_wait_asynccnt(0);
#else
  asm volatile("s_wait_asynccnt 0x0" ::: "memory");
#endif
}
#endif

// -------------------------------------------------------------------- scatter
__global__ __launch_bounds__(256) void scatter_w_kernel(
    const float* __restrict__ wvals,
    const int*   __restrict__ row,
    const int*   __restrict__ col,
    const unsigned char* __restrict__ mask,
    float* __restrict__ Wf, int nnz) {
  int i = blockIdx.x * 256 + threadIdx.x;
  if (i < nnz && mask[i]) {
    atomicAdd(Wf + (size_t)row[i] * IN_F + col[i], wvals[i]);
  }
}

// ------------------------------------------------------------------- convert
__global__ __launch_bounds__(256) void convert_bf16_kernel(
    const float* __restrict__ Wf, const float* __restrict__ x,
    bf16* __restrict__ Wb, bf16* __restrict__ Xb) {
  size_t i = ((size_t)blockIdx.x * 256 + threadIdx.x) * 4;
  float4 wf = *(const float4*)(Wf + i);
  float4 xf = *(const float4*)(x + i);
  v4bf wb = { (bf16)wf.x, (bf16)wf.y, (bf16)wf.z, (bf16)wf.w };
  v4bf xb = { (bf16)xf.x, (bf16)xf.y, (bf16)xf.z, (bf16)xf.w };
  *(v4bf*)(Wb + i) = wb;
  *(v4bf*)(Xb + i) = xb;
}

// ---------------------------------------------------------------------- GEMM
// Y[b][o] = sum_k Xb[b][k] * Wb[o][k]   (both row-major, K contiguous)
// Block: 256 threads = 8 waves (2 x 4). Wave tile 64x32 = 4x2 WMMA tiles.
// Double-buffered LDS: one barrier per K-step, tile k+1 in flight during
// tile k's WMMAs.
__global__ __launch_bounds__(256) void gemm_bf16_wmma_kernel(
    const bf16* __restrict__ X, const bf16* __restrict__ W,
    float* __restrict__ Y) {
  __shared__ alignas(16) bf16 lA[2][BM * LDSS];
  __shared__ alignas(16) bf16 lB[2][BN * LDSS];

  const int tid   = threadIdx.x;
  const int lane  = tid & 31;
  const int wave  = tid >> 5;
  const int waveM = wave >> 2;            // 0..1  -> 64-row band
  const int waveN = wave & 3;             // 0..3  -> 32-col band
  const int mBlk  = blockIdx.y * BM;
  const int nBlk  = blockIdx.x * BN;

  const v8f zero = {0.f, 0.f, 0.f, 0.f, 0.f, 0.f, 0.f, 0.f};
  v8f acc[4][2];
  for (int m = 0; m < 4; ++m)
    for (int n = 0; n < 2; ++n) acc[m][n] = zero;

  // staging: tile = 128 rows x 4 chunks of 8 bf16 (16B); 512 chunks/matrix;
  // each thread moves chunks c0 and c1 of A and of B per K-step.
  const int c0 = tid, c1 = tid + 256;
  const int r0 = c0 >> 2, e0 = (c0 & 3) * 8;   // row / element offset
  const int r1 = c1 >> 2, e1 = (c1 & 3) * 8;
  const size_t gA0 = (size_t)(mBlk + r0) * IN_F + e0;
  const size_t gA1 = (size_t)(mBlk + r1) * IN_F + e1;
  const size_t gB0 = (size_t)(nBlk + r0) * IN_F + e0;
  const size_t gB1 = (size_t)(nBlk + r1) * IN_F + e1;
  const int    l0  = r0 * LDSS + e0;
  const int    l1  = r1 * LDSS + e1;

  const int fr    = lane & 15;            // fragment row within 16
  const int half  = lane >> 4;            // 0: lanes 0-15, 1: lanes 16-31
  const int koffA = half * 8;             // A: K chunks {koffA..+7, koffA+16..+23}
  const int koffB = half * 8;             // B: K chunks {koffB..+7, koffB+16..+23}

  // ---- prologue: stage tile 0 into buffer 0
#if HAVE_ASYNC_LDS
  async_cp16(X + gA0, lA[0] + l0);
  async_cp16(X + gA1, lA[0] + l1);
  async_cp16(W + gB0, lB[0] + l0);
  async_cp16(W + gB1, lB[0] + l1);
  wait_async0();
  __syncthreads();
#else
  {
    v8bf a0 = *(const v8bf*)(X + gA0);
    v8bf a1 = *(const v8bf*)(X + gA1);
    v8bf b0 = *(const v8bf*)(W + gB0);
    v8bf b1 = *(const v8bf*)(W + gB1);
    *(v8bf*)(lA[0] + l0) = a0;
    *(v8bf*)(lA[0] + l1) = a1;
    *(v8bf*)(lB[0] + l0) = b0;
    *(v8bf*)(lB[0] + l1) = b1;
  }
  __syncthreads();
#endif

  int buf = 0;
  for (int k0 = 0; k0 < IN_F; k0 += BK) {
    const int knext = (k0 + BK) & (IN_F - 1);   // wraps to 0 on last iter (dummy)
    const int nbuf  = buf ^ 1;

    // ---- issue fetch of tile k+1 into the other buffer
#if HAVE_ASYNC_LDS
    async_cp16(X + gA0 + knext, lA[nbuf] + l0);
    async_cp16(X + gA1 + knext, lA[nbuf] + l1);
    async_cp16(W + gB0 + knext, lB[nbuf] + l0);
    async_cp16(W + gB1 + knext, lB[nbuf] + l1);
#else
    v8bf a0 = *(const v8bf*)(X + gA0 + knext);
    v8bf a1 = *(const v8bf*)(X + gA1 + knext);
    v8bf b0 = *(const v8bf*)(W + gB0 + knext);
    v8bf b1 = *(const v8bf*)(W + gB1 + knext);
#endif

    // ---- compute from current buffer
    // A fragments: 16-bit A 16x32 layout (lane halves: K 0-7/16-23, 8-15/24-31)
    v16bf afr[4];
#pragma unroll
    for (int m = 0; m < 4; ++m) {
      const bf16* p = lA[buf] + (waveM * 64 + m * 16 + fr) * LDSS + koffA;
      v8bf lo = *(const v8bf*)(p);
      v8bf hi = *(const v8bf*)(p + 16);
      afr[m] = __builtin_shufflevector(lo, hi, 0, 1, 2, 3, 4, 5, 6, 7,
                                       8, 9, 10, 11, 12, 13, 14, 15);
    }
    // B fragments: 32x16 layout (lane halves: K 0-7/16-23, 8-15/24-31, N = lane%16)
    v16bf bfr[2];
#pragma unroll
    for (int n = 0; n < 2; ++n) {
      const bf16* p = lB[buf] + (waveN * 32 + n * 16 + fr) * LDSS + koffB;
      v8bf lo = *(const v8bf*)(p);
      v8bf hi = *(const v8bf*)(p + 16);
      bfr[n] = __builtin_shufflevector(lo, hi, 0, 1, 2, 3, 4, 5, 6, 7,
                                       8, 9, 10, 11, 12, 13, 14, 15);
    }

#pragma unroll
    for (int m = 0; m < 4; ++m)
#pragma unroll
      for (int n = 0; n < 2; ++n)
        acc[m][n] = __builtin_amdgcn_wmma_f32_16x16x32_bf16(
            false, afr[m], false, bfr[n], (short)0, acc[m][n], false, false);

    // ---- complete staging of tile k+1, then one barrier
#if HAVE_ASYNC_LDS
    wait_async0();
#else
    *(v8bf*)(lA[nbuf] + l0) = a0;
    *(v8bf*)(lA[nbuf] + l1) = a1;
    *(v8bf*)(lB[nbuf] + l0) = b0;
    *(v8bf*)(lB[nbuf] + l1) = b1;
#endif
    __syncthreads();
    buf = nbuf;
  }

  // Epilogue: C/D f32 layout — VGPR r <-> M=r (+8 for lane half 1), N = lane%16
  const int mo = half << 3;
#pragma unroll
  for (int m = 0; m < 4; ++m)
#pragma unroll
    for (int n = 0; n < 2; ++n) {
      const int gcol = nBlk + waveN * 32 + n * 16 + fr;
#pragma unroll
      for (int r = 0; r < 8; ++r) {
        const int grow = mBlk + waveM * 64 + m * 16 + r + mo;
        Y[(size_t)grow * OUT_F + gcol] = acc[m][n][r];
      }
    }
}

// -------------------------------------------------------------------- launch
extern "C" void kernel_launch(void* const* d_in, const int* in_sizes, int n_in,
                              void* d_out, int out_size, void* d_ws, size_t ws_size,
                              hipStream_t stream) {
  const float* x     = (const float*)d_in[0];
  const float* wvals = (const float*)d_in[1];
  const int*   idx   = (const int*)d_in[2];
  const unsigned char* mask = (const unsigned char*)d_in[3];
  const int nnz = in_sizes[1];
  const int* row = idx;
  const int* col = idx + nnz;

  // workspace: [W_f32 64MB][W_bf16 32MB][X_bf16 32MB]
  float* Wf = (float*)d_ws;
  bf16*  Wb = (bf16*)((char*)d_ws + (size_t)OUT_F * IN_F * 4);
  bf16*  Xb = (bf16*)((char*)d_ws + (size_t)OUT_F * IN_F * 4
                                  + (size_t)OUT_F * IN_F * 2);
  float* Y = (float*)d_out;

  (void)hipMemsetAsync(Wf, 0, (size_t)OUT_F * IN_F * sizeof(float), stream);

  scatter_w_kernel<<<(nnz + 255) / 256, 256, 0, stream>>>(
      wvals, row, col, mask, Wf, nnz);

  const int nconv = (OUT_F * IN_F) / 4;   // 4 elems per thread
  convert_bf16_kernel<<<nconv / 256, 256, 0, stream>>>(Wf, x, Wb, Xb);

  dim3 grid(OUT_F / BN, BATCH / BM);
  gemm_bf16_wmma_kernel<<<grid, 256, 0, stream>>>(Xb, Wb, Y);
}
